// _Query_Guided_Attention_1151051235525
// MI455X (gfx1250) — matrix operand order repaired and checked by reading, not compiled, not run on hardware
//
#include <hip/hip_runtime.h>
#include <hip/hip_fp16.h>
#include <math.h>

typedef __attribute__((ext_vector_type(16))) _Float16 v16h;
typedef __attribute__((ext_vector_type(8)))  _Float16 v8h;
typedef __attribute__((ext_vector_type(8)))  float    v8f;
typedef __attribute__((ext_vector_type(4)))  unsigned int v4u;
typedef __attribute__((ext_vector_type(4)))  int v4i;

typedef __attribute__((address_space(1))) v4i* gas_v4i_p;   // global
typedef __attribute__((address_space(3))) v4i* las_v4i_p;   // LDS

#define BATCH 16
#define C_IN  256
#define C_MID 128
#define HW    4096   // 64*64
#define XSTRIDE 136  // proj LDS row stride (halves)

// Phase-2 tiling
#define JT      64           // j-columns staged per LDS buffer
#define NT      (HW / JT)    // 64 tiles
#define PSTRIDE 136          // 128 + 8 pad halves -> conflict-free ds_load_b128

#if __has_builtin(__builtin_amdgcn_global_load_async_to_lds_b128)
#define USE_ASYNC_LDS 1
#else
#define USE_ASYNC_LDS 0
#endif

#if __has_builtin(__builtin_amdgcn_s_wait_asynccnt)
#define WAIT_ASYNC(n) __builtin_amdgcn_s_wait_asynccnt(n)
#else
#define WAIT_ASYNC(n) asm volatile("s_wait_asynccnt %0" :: "i"(n) : "memory")
#endif

// -----------------------------------------------------------------------------
// Phase 1: projection  out[b, i, o] = sum_c x[b, c, i] * w[o, c] + bias[o]
// -----------------------------------------------------------------------------
__global__ __launch_bounds__(256) void proj_kernel(
    const float* __restrict__ x,      // [B, C_IN, HW]
    const float* __restrict__ w,      // [C_MID, C_IN]
    const float* __restrict__ bias,   // [C_MID]
    _Float16*    __restrict__ out)    // [B, HW, C_MID]
{
  __shared__ __align__(16) _Float16 xs[16 * XSTRIDE];

  const int b     = blockIdx.x >> 5;           // 32 blocks per batch
  const int tile0 = (blockIdx.x & 31) * 128;   // 128 pixels per block
  const int wid   = threadIdx.x >> 5;
  const int lane  = threadIdx.x & 31;
  const int lrow  = lane & 15;
  const int lhi   = lane >> 4;
  const int ocol  = wid * 16 + lrow;

  // Hoisted weight B-fragments: B[k=c, n=o] = w[o, c]; lane K = kstep*32 + 8*lhi + (h & 7) + 16*(h >> 3)
  v16h bw[8];
  #pragma unroll
  for (int k = 0; k < 8; ++k) {
    const float* wp = w + (size_t)ocol * C_IN + k * 32 + lhi * 8;
    #pragma unroll
    for (int t = 0; t < 16; ++t) bw[k][t] = (_Float16)wp[t + (t & 8)];
  }
  const float bv = bias[ocol];

  for (int t = 0; t < 8; ++t) {
    const int pix0 = tile0 + t * 16;
    #pragma unroll
    for (int r = 0; r < 16; ++r) {
      const int e = threadIdx.x + r * 256;
      const int c = e >> 4;
      const int i = e & 15;
      xs[i * XSTRIDE + c] = (_Float16)x[((size_t)b * C_IN + c) * HW + pix0 + i];
    }
    __syncthreads();

    v8f acc = {};
    #pragma unroll
    for (int k = 0; k < 8; ++k) {
      const int kbase = k * 32 + lhi * 8;
      const _Float16* ap = &xs[lrow * XSTRIDE + kbase];
      v8h lo = *(const v8h*)(ap);
      v8h hi = *(const v8h*)(ap + 16);
      v16h a;
      #pragma unroll
      for (int q = 0; q < 8; ++q) { a[q] = lo[q]; a[q + 8] = hi[q]; }
      acc = __builtin_amdgcn_wmma_f32_16x16x32_f16(
          false, a, false, bw[k], (short)0, acc, false, false);
    }

    #pragma unroll
    for (int v = 0; v < 8; ++v) {
      const int i = pix0 + v + 8 * lhi;
      out[((size_t)b * HW + i) * C_MID + ocol] = (_Float16)(acc[v] + bv);
    }
    __syncthreads();
  }
}

// -----------------------------------------------------------------------------
// Phase-2 staging helpers: copy halves 0..63 of 128 rows of P into padded LDS.
// Each of 256 threads moves 4 x 16B chunks.
// -----------------------------------------------------------------------------
#if USE_ASYNC_LDS
__device__ __forceinline__ void stage_async(const _Float16* gsrc, _Float16* lbuf, int tid) {
  #pragma unroll
  for (int p = 0; p < 4; ++p) {
    const int c   = p * 256 + tid;
    const int row = c >> 3;
    const int seg = c & 7;
    __builtin_amdgcn_global_load_async_to_lds_b128(
        (gas_v4i_p)(v4i*)(gsrc + row * C_MID + seg * 8),
        (las_v4i_p)(v4i*)(lbuf + row * PSTRIDE + seg * 8),
        0, 0);
  }
}
#else
__device__ __forceinline__ void prefetch_regs(const _Float16* gsrc, v4u r[4], int tid) {
  #pragma unroll
  for (int p = 0; p < 4; ++p) {
    const int c   = p * 256 + tid;
    const int row = c >> 3;
    const int seg = c & 7;
    r[p] = *(const v4u*)(gsrc + row * C_MID + seg * 8);
  }
}
__device__ __forceinline__ void store_regs(const v4u r[4], _Float16* lbuf, int tid) {
  #pragma unroll
  for (int p = 0; p < 4; ++p) {
    const int c   = p * 256 + tid;
    const int row = c >> 3;
    const int seg = c & 7;
    *(v4u*)(lbuf + row * PSTRIDE + seg * 8) = r[p];
  }
}
#endif

// -----------------------------------------------------------------------------
// Phase 2: out[b,i] = sigmoid(max_j <T[i,:], P[j,:]> / 4096)
// Block: 8 waves x 32 rows = 256 rows. P tiles double-buffered in LDS.
// -----------------------------------------------------------------------------
__global__ __launch_bounds__(256) void attn_kernel(
    const _Float16* __restrict__ T,   // [B, HW, C_MID]
    const _Float16* __restrict__ P,   // [B, HW, C_MID]
    float*          __restrict__ out) // [B, HW]
{
  __shared__ __align__(16) _Float16 pbuf[2][JT * PSTRIDE];

  const int tid  = threadIdx.x;
  const int b    = blockIdx.x >> 4;                       // 16 blocks per batch
  const int wid  = tid >> 5;
  const int lane = tid & 31;
  const int lrow = lane & 15;
  const int lhi  = lane >> 4;
  const int i0   = (blockIdx.x & 15) * 256 + wid * 32;    // 32 rows per wave

  const _Float16* Tb = T + (size_t)b * HW * C_MID;
  const _Float16* Pb = P + (size_t)b * HW * C_MID;

  // Two hoisted A-fragment sets (rows i0..i0+15 and i0+16..i0+31)
  v16h a0[4], a1[4];
  #pragma unroll
  for (int k = 0; k < 4; ++k) {
    const int kbase = k * 32 + lhi * 8;
    const _Float16* ap0 = Tb + (size_t)(i0 + lrow) * C_MID + kbase;
    const _Float16* ap1 = ap0 + 16 * C_MID;
    v8h lo0 = *(const v8h*)(ap0), hi0 = *(const v8h*)(ap0 + 16);
    v8h lo1 = *(const v8h*)(ap1), hi1 = *(const v8h*)(ap1 + 16);
    #pragma unroll
    for (int q = 0; q < 8; ++q) {
      a0[k][q] = lo0[q]; a0[k][q + 8] = hi0[q];
      a1[k][q] = lo1[q]; a1[k][q + 8] = hi1[q];
    }
  }

  float rmax0[8], rmax1[8];
  #pragma unroll
  for (int v = 0; v < 8; ++v) { rmax0[v] = -INFINITY; rmax1[v] = -INFINITY; }

#if USE_ASYNC_LDS
  stage_async(Pb, &pbuf[0][0], tid);
#else
  v4u pr[4];
  prefetch_regs(Pb, pr, tid);
#endif

  for (int jt = 0; jt < NT; ++jt) {
    const int cur = jt & 1;

#if USE_ASYNC_LDS
    if (jt > 0) __syncthreads();                      // prior compute done before refilling other buf
    if (jt + 1 < NT) {
      stage_async(Pb + (size_t)(jt + 1) * JT * C_MID, &pbuf[1 - cur][0], tid);
      WAIT_ASYNC(4);                                  // current buffer's 4 loads done
    } else {
      WAIT_ASYNC(0);
    }
    __syncthreads();                                  // all waves' copies of buf[cur] done
#else
    __syncthreads();                                  // prior compute on buf[cur] done
    store_regs(pr, &pbuf[cur][0], tid);
    if (jt + 1 < NT) prefetch_regs(Pb + (size_t)(jt + 1) * JT * C_MID, pr, tid);
    __syncthreads();                                  // buf[cur] fully written
#endif

    // Consume buf[cur]: 4 sub-tiles of 16 columns, K = 128 = 4 WMMA steps
    #pragma unroll
    for (int js = 0; js < 4; ++js) {
      v8f acc0 = {}, acc1 = {};
      #pragma unroll
      for (int k = 0; k < 4; ++k) {
        const _Float16* bp = &pbuf[cur][(js * 16 + lrow) * PSTRIDE + k * 32 + lhi * 8];
        v8h lo = *(const v8h*)(bp);
        v8h hi = *(const v8h*)(bp + 16);
        v16h bf;
        #pragma unroll
        for (int q = 0; q < 8; ++q) { bf[q] = lo[q]; bf[q + 8] = hi[q]; }
        acc0 = __builtin_amdgcn_wmma_f32_16x16x32_f16(
            false, a0[k], false, bf, (short)0, acc0, false, false);
        acc1 = __builtin_amdgcn_wmma_f32_16x16x32_f16(
            false, a1[k], false, bf, (short)0, acc1, false, false);
      }
      #pragma unroll
      for (int v = 0; v < 8; ++v) {
        rmax0[v] = fmaxf(rmax0[v], acc0[v]);
        rmax1[v] = fmaxf(rmax1[v], acc1[v]);
      }
    }
  }

  // Row-max across the 16 lanes sharing each row set, then sigmoid(max/4096)
  #pragma unroll
  for (int v = 0; v < 8; ++v) {
    float m0 = rmax0[v], m1 = rmax1[v];
    m0 = fmaxf(m0, __shfl_xor(m0, 1, 32)); m1 = fmaxf(m1, __shfl_xor(m1, 1, 32));
    m0 = fmaxf(m0, __shfl_xor(m0, 2, 32)); m1 = fmaxf(m1, __shfl_xor(m1, 2, 32));
    m0 = fmaxf(m0, __shfl_xor(m0, 4, 32)); m1 = fmaxf(m1, __shfl_xor(m1, 4, 32));
    m0 = fmaxf(m0, __shfl_xor(m0, 8, 32)); m1 = fmaxf(m1, __shfl_xor(m1, 8, 32));
    rmax0[v] = m0; rmax1[v] = m1;
  }
  if (lrow == 0) {
    #pragma unroll
    for (int v = 0; v < 8; ++v) {
      const float f0 = rmax0[v] * (1.0f / 4096.0f);
      const float f1 = rmax1[v] * (1.0f / 4096.0f);
      out[(size_t)b * HW + i0 + v + 8 * lhi]      = 1.0f / (1.0f + expf(-f0));
      out[(size_t)b * HW + i0 + 16 + v + 8 * lhi] = 1.0f / (1.0f + expf(-f1));
    }
  }
}

// -----------------------------------------------------------------------------
extern "C" void kernel_launch(void* const* d_in, const int* in_sizes, int n_in,
                              void* d_out, int out_size, void* d_ws, size_t ws_size,
                              hipStream_t stream) {
  const float* x_gallery = (const float*)d_in[0];
  const float* x_query   = (const float*)d_in[1];
  const float* theta_w   = (const float*)d_in[2];
  const float* theta_b   = (const float*)d_in[3];
  const float* phi_w     = (const float*)d_in[4];
  const float* phi_b     = (const float*)d_in[5];
  float* out = (float*)d_out;

  _Float16* T = (_Float16*)d_ws;                         // 16 MB
  _Float16* P = T + (size_t)BATCH * HW * C_MID;          // 16 MB

  proj_kernel<<<dim3(BATCH * (HW / 128)), dim3(256), 0, stream>>>(x_gallery, theta_w, theta_b, T);
  proj_kernel<<<dim3(BATCH * (HW / 128)), dim3(256), 0, stream>>>(x_query,   phi_w,   phi_b,   P);
  attn_kernel<<<dim3(BATCH * (HW / 256)), dim3(256), 0, stream>>>(T, P, out);
}
